// KinspeakEmformerRNNT_62912680952346
// MI455X (gfx1250) — compile-verified, matrix operand order checked by reading, not run on hardware
//
#include <hip/hip_runtime.h>
#include <hip/hip_bf16.h>

typedef __attribute__((ext_vector_type(16))) _Float16 v16h;
typedef __attribute__((ext_vector_type(8)))  float    v8f;

#define RB 8
#define RT 128
#define RU 100
#define RV 1024
#define NEGF (-1.0e30f)

// ---------------------------------------------------------------------------
// Kernel 1: per-row logsumexp over V=1024, then emit blank/target log-probs.
// One wave32 per row; 32 f32 per lane held in registers (single HBM pass).
// Row sum via v_wmma_f32_16x16x32_f16 against an all-ones B matrix:
// B==1 makes the K-layout irrelevant, and since every lane of the wave holds
// values of the SAME logical row, sum_m D[m,n] == full row sum (f32 accum).
// ---------------------------------------------------------------------------
__global__ __launch_bounds__(256) void rnnt_lse_kernel(
    const float* __restrict__ logits,
    const int*   __restrict__ targets,
    float* __restrict__ blank_lp,   // [B*T*(U+1)]
    float* __restrict__ emit_lp)    // [B*T*U]
{
    const int lane = threadIdx.x & 31;
    const int wave = threadIdx.x >> 5;
    const int row  = blockIdx.x * 8 + wave;          // exact: grid = rows/8
    const size_t base = (size_t)row * RV;

    // Coalesced: lane reads float4s at stride 32 -> 32 f32 per lane.
    const float4* p = (const float4*)(logits + base);
    float4 x[8];
#pragma unroll
    for (int j = 0; j < 8; ++j) x[j] = p[lane + 32 * j];

    // Row max: lane-local then wave32 butterfly reduce.
    float m = -3.402823466e38f;
#pragma unroll
    for (int j = 0; j < 8; ++j) {
        m = fmaxf(m, fmaxf(fmaxf(x[j].x, x[j].y), fmaxf(x[j].z, x[j].w)));
    }
#pragma unroll
    for (int off = 16; off > 0; off >>= 1) m = fmaxf(m, __shfl_xor(m, off, 32));

    // exp(x - max) in f32, pack to f16 for the WMMA reduction (values <= 1).
    float e[32];
#pragma unroll
    for (int j = 0; j < 8; ++j) {
        e[4 * j + 0] = __expf(x[j].x - m);
        e[4 * j + 1] = __expf(x[j].y - m);
        e[4 * j + 2] = __expf(x[j].z - m);
        e[4 * j + 3] = __expf(x[j].w - m);
    }
    v16h a0, a1, ones;
#pragma unroll
    for (int k = 0; k < 16; ++k) {
        a0[k]   = (_Float16)e[k];
        a1[k]   = (_Float16)e[16 + k];
        ones[k] = (_Float16)1.0f;
    }
    // D = A * Ones + C : every D[m,n] = partial sum of this row's values.
    v8f c = {};
    c = __builtin_amdgcn_wmma_f32_16x16x32_f16(false, a0, false, ones,
                                               (short)0, c, false, false);
    c = __builtin_amdgcn_wmma_f32_16x16x32_f16(false, a1, false, ones,
                                               (short)0, c, false, false);
    float s = c[0] + c[1] + c[2] + c[3] + c[4] + c[5] + c[6] + c[7];
    s += __shfl_xor(s, 16, 32);                      // half-wave partials -> total
    const float lse = m + __logf(s);

    // Decode (b, t, u) and write the two consumed log-probs.
    const int b   = row / (RT * (RU + 1));
    const int rem = row % (RT * (RU + 1));
    const int t   = rem / (RU + 1);
    const int u   = rem % (RU + 1);
    if (lane == 0) {
        blank_lp[row] = logits[base] - lse;          // v = BLANK = 0
    }
    if (lane == 1 && u < RU) {
        const int tgt = targets[b * RU + u];
        emit_lp[(b * RT + t) * RU + u] = logits[base + tgt] - lse;
    }
}

// ---------------------------------------------------------------------------
// Kernel 2: per-batch alpha DP over anti-diagonals, fully LDS-staged.
// alpha[t][u] = logaddexp(alpha[t-1][u] + blank[t-1][u],
//                         alpha[t][u-1] + emit[t][u-1])
// Only the previous diagonal is kept (cell (t,u) depends on (t-1,u),(t,u-1),
// both on diagonal d-1, indexed by t).
// ---------------------------------------------------------------------------
__global__ __launch_bounds__(128) void rnnt_dp_kernel(
    const float* __restrict__ blank_lp,
    const float* __restrict__ emit_lp,
    const int*   __restrict__ logit_lengths,
    const int*   __restrict__ target_lengths,
    float* __restrict__ nll)                         // [B]
{
    extern __shared__ float smem[];
    float* blankS = smem;                            // T*(U+1)
    float* emitS  = blankS + RT * (RU + 1);          // T*U
    float* diag0  = emitS + RT * RU;                 // 128
    float* diag1  = diag0 + 128;                     // 128

    const int b   = blockIdx.x;
    const int tid = threadIdx.x;

    // Stage this batch's tables into LDS (each value read exactly once later).
    const float* gb = blank_lp + (size_t)b * RT * (RU + 1);
    for (int i = tid; i < RT * (RU + 1); i += blockDim.x) blankS[i] = gb[i];
    const float* ge = emit_lp + (size_t)b * RT * RU;
    for (int i = tid; i < RT * RU; i += blockDim.x) emitS[i] = ge[i];
    diag0[tid] = (tid == 0) ? 0.0f : NEGF;           // alpha[0][0] = 0
    __syncthreads();

    const int tl   = logit_lengths[b] - 1;           // reference: t_len
    const int ul   = target_lengths[b];
    const int tfin = tl - 1;                         // reference indexes t_len-1
    const int dmax = tfin + ul;

    float* prev = diag0;
    float* cur  = diag1;
    float  ans  = 0.0f;

    for (int d = 1; d <= dmax; ++d) {
        const int t = tid;
        const int u = d - t;
        if (u >= 0 && u <= RU && t < RT) {
            const float up = (t > 0) ? prev[t - 1] + blankS[(t - 1) * (RU + 1) + u]
                                     : NEGF;
            const float lf = (u > 0) ? prev[t] + emitS[t * RU + (u - 1)]
                                     : NEGF;
            const float mx  = fmaxf(up, lf);
            const float mn  = fminf(up, lf);
            const float val = mx + log1pf(__expf(mn - mx));   // logaddexp
            cur[t] = val;
            if (d == dmax && t == tfin) ans = val;
        }
        __syncthreads();
        float* tmp = prev; prev = cur; cur = tmp;
    }

    if (tid == tfin) {
        nll[b] = -(ans + blankS[tfin * (RU + 1) + ul]);
    }
}

// ---------------------------------------------------------------------------
// Kernel 3: mean over B nll values -> scalar output.
// ---------------------------------------------------------------------------
__global__ void rnnt_mean_kernel(const float* __restrict__ nll,
                                 float* __restrict__ out)
{
    const int lane = threadIdx.x;
    float v = (lane < RB) ? nll[lane] : 0.0f;
#pragma unroll
    for (int off = 16; off > 0; off >>= 1) v += __shfl_xor(v, off, 32);
    if (lane == 0) out[0] = v / (float)RB;
}

extern "C" void kernel_launch(void* const* d_in, const int* in_sizes, int n_in,
                              void* d_out, int out_size, void* d_ws, size_t ws_size,
                              hipStream_t stream) {
    (void)in_sizes; (void)n_in; (void)out_size; (void)ws_size;
    const float* logits         = (const float*)d_in[0];
    const int*   targets        = (const int*)d_in[1];
    const int*   logit_lengths  = (const int*)d_in[2];
    const int*   target_lengths = (const int*)d_in[3];

    float* ws       = (float*)d_ws;
    float* blank_lp = ws;                                  // B*T*(U+1)
    float* emit_lp  = blank_lp + RB * RT * (RU + 1);       // B*T*U
    float* nll      = emit_lp + RB * RT * RU;              // B

    const int rows = RB * RT * (RU + 1);                   // 103424, /8 exact
    rnnt_lse_kernel<<<rows / 8, 256, 0, stream>>>(logits, targets,
                                                  blank_lp, emit_lp);

    const size_t shmem = (size_t)(RT * (RU + 1) + RT * RU + 2 * 128) * sizeof(float);
    rnnt_dp_kernel<<<RB, 128, shmem, stream>>>(blank_lp, emit_lp,
                                               logit_lengths, target_lengths, nll);

    rnnt_mean_kernel<<<1, 32, 0, stream>>>(nll, (float*)d_out);
}
